// DeepTensorNN_66211215835150
// MI455X (gfx1250) — matrix operand order repaired and checked by reading, not compiled, not run on hardware
//
#include <hip/hip_runtime.h>

typedef __attribute__((ext_vector_type(16))) _Float16 v16h;
typedef __attribute__((ext_vector_type(8)))  float    v8f;

#define BATCH 64
#define NATOM 256
#define AEMB  20
#define NRBF  25
#define NF    45   // AEMB + NRBF

// exp(x) and tanh(x) on the transcendental pipe (v_exp2 / v_rcp), no libcalls.
__device__ __forceinline__ float fast_tanh(float x) {
  // tanh(x) = 1 - 2/(exp(2x)+1);  exp(2x) = exp2(2*log2(e)*x)
  float e = __builtin_amdgcn_exp2f(x * 2.885390081777927f);
  return 1.0f - 2.0f * __builtin_amdgcn_rcpf(e + 1.0f);
}

// One wave32 per (b,i) atom.  8 waves / 256-thread block.
__global__ void __launch_bounds__(256)
dtnn_kernel(const int* __restrict__ z, const float* __restrict__ dist,
            const float* __restrict__ emb, const float* __restrict__ Vw,
            const float* __restrict__ Vb, const float* __restrict__ W1,
            const float* __restrict__ b1, const float* __restrict__ W2,
            const float* __restrict__ b2,
            float* __restrict__ pair_out,   // [BATCH*NATOM] per-atom energies (ws path)
            float* __restrict__ atom_out)   // [BATCH] (atomic fallback path)
{
  const int lane = threadIdx.x & 31;
  const int wid  = threadIdx.x >> 5;
  const int pair = blockIdx.x * 8 + wid;      // 0 .. BATCH*NATOM-1
  const int b = pair >> 8;
  const int i = pair & (NATOM - 1);

  const int  nB = lane & 15;
  const bool hi = lane >= 16;

  // ---------------- per-atom (wave-uniform) setup ----------------
  const int   zi    = z[b * NATOM + i];
  const float maskf = (zi != 0) ? 1.0f : 0.0f;

  // Column indices this lane owns in the two output tiles: o0 = nB, o1 = 16+nB
  const int o1c = (nB < 4) ? (16 + nB) : (AEMB - 1);   // clamped for safe loads

  // a = cfeat . Vw[:, :20] + Vb  (j-independent part of the message pre-activation)
  float a0 = Vb[nB];
  float a1 = Vb[o1c];
  const int oc = (lane < AEMB) ? lane : 0;
  float cf_lane = emb[zi * AEMB + oc] * maskf;         // cfeat[lane] (valid lane<20)
  float w_eff = 0.0f;                                  // sum_p W2[p]*W1[p, lane]
#pragma unroll
  for (int p = 0; p < 10; ++p) w_eff += W2[p] * W1[p * AEMB + oc];
#pragma unroll
  for (int f = 0; f < AEMB; ++f) {
    float cf = emb[zi * AEMB + f] * maskf;
    a0 += cf * Vw[nB  * NF + f];
    a1 += cf * Vw[o1c * NF + f];
  }

  // ---------------- B tiles: Vw[:,20:45] as 32x16 f16 (K padded 25->32) ------
  // B layout used here (contiguous k, same order as A): lanes 0-15 hold K=0..15, lanes 16-31 hold K=16..31,
  // column N = lane%16; v16h element t -> K = kbase + t.
  const int kbaseB = hi ? 16 : 0;
  v16h B0, B1;
#pragma unroll
  for (int t = 0; t < 16; ++t) {
    int k  = kbaseB + t;
    int kc = (k < NRBF) ? k : (NRBF - 1);
    float v0 = Vw[nB  * NF + AEMB + kc];
    float v1 = Vw[o1c * NF + AEMB + kc];
    v0 = (k < NRBF) ? v0 : 0.0f;
    v1 = (k < NRBF && nB < 4) ? v1 : 0.0f;   // tile1 has only 4 valid columns
    B0[t] = (_Float16)v0;
    B1[t] = (_Float16)v1;
  }

  // ---------------- main loop over neighbor tiles (16 j per tile) ------------
  // A layout used here (16x32, contiguous k, same order as B): lane M = lane%16;
// lanes 0-15: K = 0..15 (elems 0-15)
// lanes 16-31: K = 16..31  ->  k = (hi?16:0) + t
  const float* drow  = dist + ((size_t)(b * NATOM + i)) * NATOM;
  const int    abase = hi ? 16 : 0;
  const v8f    czero = {};
  float acc0 = 0.0f, acc1 = 0.0f;

  for (int jt = 0; jt < NATOM / 16; ++jt) {
    float d = __builtin_nontemporal_load(drow + jt * 16 + nB);  // streamed once
    v16h A;
#pragma unroll
    for (int t = 0; t < 16; ++t) {
      int   k  = abase + t;
      float mu = 0.2f * (float)k;
      float tt = d - mu;
      // exp(-2*tt^2) = exp2(-2*log2(e)*tt^2)
      float g  = __builtin_amdgcn_exp2f(-2.885390081777927f * (tt * tt));
      g = (k < NRBF) ? g : 0.0f;
      A[t] = (_Float16)g;
    }
    v8f d0 = __builtin_amdgcn_wmma_f32_16x16x32_f16(false, A, false, B0,
                                                    (short)0, czero, false, false);
    v8f d1 = __builtin_amdgcn_wmma_f32_16x16x32_f16(false, A, false, B1,
                                                    (short)0, czero, false, false);
#pragma unroll
    for (int r = 0; r < 8; ++r) {
      acc0 += fast_tanh(d0[r] + a0);   // msg for column nB,    rows M=r(+8 if hi)
      acc1 += fast_tanh(d1[r] + a1);   // msg for column 16+nB
    }
  }
  // i-mask applied after tanh (sum over j is linear in msg)
  acc0 *= maskf;
  acc1 *= maskf;

  // Combine lane halves: full sum over the 16 M rows for this lane's column.
  float agg0 = acc0 + __shfl_xor(acc0, 16, 32);
  float agg1 = acc1 + __shfl_xor(acc1, 16, 32);

  // ---------------- collapsed top MLP -----------------
  // lane o (<20): c_o = cfeat[o] + agg[o];  e = econst + sum_o tanh(c_o)*w_eff[o]
  float aggo = (lane < 16) ? agg0 : agg1;          // lanes 16..19 hold cols 16..19
  float th   = fast_tanh(cf_lane + aggo);
  float contrib = (lane < AEMB) ? th * w_eff : 0.0f;
#pragma unroll
  for (int off = 16; off >= 1; off >>= 1)
    contrib += __shfl_xor(contrib, off, 32);

  if (lane == 0) {
    float econst = b2[0];
#pragma unroll
    for (int p = 0; p < 10; ++p) econst += W2[p] * b1[p];
    float e = contrib + econst;
    if (pair_out) pair_out[pair] = e;          // deterministic path
    else          atomicAdd(atom_out + b, e);  // fallback
  }
}

// Deterministic per-batch reduction of the 256 per-atom energies.
__global__ void __launch_bounds__(256)
reduce_kernel(const float* __restrict__ ws, float* __restrict__ out) {
  __shared__ float sm[256];
  const int b = blockIdx.x;
  sm[threadIdx.x] = ws[b * NATOM + threadIdx.x];
  __syncthreads();
  for (int s = 128; s > 0; s >>= 1) {
    if (threadIdx.x < s) sm[threadIdx.x] += sm[threadIdx.x + s];
    __syncthreads();
  }
  if (threadIdx.x == 0) out[b] = sm[0];
}

__global__ void zero_out_kernel(float* out) {
  if (threadIdx.x < BATCH) out[threadIdx.x] = 0.0f;
}

extern "C" void kernel_launch(void* const* d_in, const int* in_sizes, int n_in,
                              void* d_out, int out_size, void* d_ws, size_t ws_size,
                              hipStream_t stream) {
  const int*   z    = (const int*)  d_in[0];
  const float* dist = (const float*)d_in[1];
  const float* emb  = (const float*)d_in[2];
  const float* Vw   = (const float*)d_in[3];
  const float* Vb   = (const float*)d_in[4];
  const float* W1   = (const float*)d_in[5];
  const float* b1   = (const float*)d_in[6];
  const float* W2   = (const float*)d_in[7];
  const float* b2   = (const float*)d_in[8];
  float* out = (float*)d_out;

  const int nblocks = (BATCH * NATOM) / 8;   // 8 waves (one atom each) per block
  const bool use_ws = d_ws && ws_size >= (size_t)(BATCH * NATOM) * sizeof(float);

  if (use_ws) {
    float* pair_ws = (float*)d_ws;
    dtnn_kernel<<<nblocks, 256, 0, stream>>>(z, dist, emb, Vw, Vb, W1, b1, W2, b2,
                                             pair_ws, nullptr);
    reduce_kernel<<<BATCH, 256, 0, stream>>>(pair_ws, out);
  } else {
    zero_out_kernel<<<1, 64, 0, stream>>>(out);
    dtnn_kernel<<<nblocks, 256, 0, stream>>>(z, dist, emb, Vw, Vb, W1, b1, W2, b2,
                                             nullptr, out);
  }
}
